// DifferentiableSpiralRenderer_12678743458084
// MI455X (gfx1250) — matrix operand order repaired and checked by reading, not compiled, not run on hardware
//
#include <hip/hip_runtime.h>

typedef __attribute__((ext_vector_type(16))) _Float16 v16h;
typedef __attribute__((ext_vector_type(8)))  float    v8f;

#define IMG_H 224
#define IMG_W 224
#define NSPLAT 381
#define KCHUNK 32
#define NPAD   384          // NSPLAT padded to multiple of KCHUNK (12 chunks)
#define NBATCH 16
#define TILE 16             // pixels per wave (one WMMA M-tile)
#define WAVES_PER_BLOCK 8   // 256 threads, wave32
#define TILES_PER_IMG ((IMG_H * IMG_W) / TILE)              // 3136
#define BLOCKS_PER_IMG (TILES_PER_IMG / WAVES_PER_BLOCK)    // 392
#define NCOL 16             // B/D matrix width (channels 0..2 live, rest zero)
#define COLSTRIDE 392       // halfs per color row: 196 dwords, 196%64=4 -> lane n
                            // starts at bank 4n, 16 lanes tile 64 banks, 0 conflicts

static_assert(IMG_W % TILE == 0, "tile must not cross image rows");
static_assert(TILES_PER_IMG % WAVES_PER_BLOCK == 0, "blocks per image must be exact");
static_assert(NPAD % KCHUNK == 0, "padded splat count must be chunk multiple");
static_assert(COLSTRIDE >= NPAD + 8, "color row must cover padded K range");

__global__ __launch_bounds__(256)
void splat_wmma_kernel(const float* __restrict__ positions, // [B,N,2]
                       const float* __restrict__ colors,    // [B,N,3]
                       const float* __restrict__ sizes,     // [B,N]
                       float* __restrict__ out)             // [B,H,W,3]
{
    // Zero-padded per-frame splat data: no bounds checks anywhere in the k-loop.
    __shared__ __align__(16) float    s_px[NPAD];
    __shared__ __align__(16) float    s_py[NPAD];
    __shared__ __align__(16) float    s_ns[NPAD];                // -log2(e)/(2*sigma^2+1e-8); 0 in pad
    __shared__ __align__(16) _Float16 s_colT[NCOL * COLSTRIDE];  // [col][k], zero outside (n<3, k<NSPLAT)

    const int tid          = threadIdx.x;
    const int b            = blockIdx.x / BLOCKS_PER_IMG;
    const int block_in_img = blockIdx.x % BLOCKS_PER_IMG;

    for (int i = tid; i < NPAD; i += 256) {
        float px = 0.0f, py = 0.0f, ns = 0.0f;   // pad: weight=exp2(0)=1, color=0 -> contributes 0
        if (i < NSPLAT) {
            px = positions[((size_t)b * NSPLAT + i) * 2 + 0];
            py = positions[((size_t)b * NSPLAT + i) * 2 + 1];
            const float sz    = sizes[(size_t)b * NSPLAT + i];
            const float sigma = sz * (2.0f / (float)IMG_H);
            ns = -1.44269504088896340736f / (2.0f * sigma * sigma + 1e-8f);
        }
        s_px[i] = px; s_py[i] = py; s_ns[i] = ns;
    }
    for (int i = tid; i < NCOL * COLSTRIDE; i += 256) {
        const int nn = i / COLSTRIDE;
        const int k  = i % COLSTRIDE;
        const float c = (nn < 3 && k < NSPLAT)
                      ? colors[((size_t)b * NSPLAT + k) * 3 + nn] : 0.0f;
        s_colT[i] = (_Float16)c;
    }
    __syncthreads();

    const int wave = tid >> 5;          // wave32
    const int lane = tid & 31;
    const int tile = block_in_img * WAVES_PER_BLOCK + wave;  // 0..3135
    const int p0   = tile * TILE;
    const int row  = p0 / IMG_W;
    const int col0 = p0 % IMG_W;

    const int m    = lane & 15;         // A-matrix row = pixel within tile
    const int half = lane >> 4;         // lane group (K split per ISA 7.12.2)
    const int n    = lane & 15;         // B/D column = color channel

    const float gx = -1.0f + (2.0f / (float)(IMG_W - 1)) * (float)(col0 + m);
    const float gy = -1.0f + (2.0f / (float)(IMG_H - 1)) * (float)row;

    v8f acc = {};

#pragma unroll
    for (int kk = 0; kk < NPAD; kk += KCHUNK) {
        // ---- A fragment: f16 Gaussian weights (wave-uniform straight-line code) ----
        v16h a;
#pragma unroll
        for (int e = 0; e < 16; ++e) {
            const int k = kk + half * 16 + e;
            const float dx = gx - s_px[k];
            const float dy = gy - s_py[k];
            const float d2 = dx * dx + dy * dy;
            // v_exp_f32 is 2^x; scale pre-folded into s_ns -> exactly mul+exp
            a[e] = (_Float16)__builtin_amdgcn_exp2f(d2 * s_ns[k]);
        }
        // ---- B fragment: 16 contiguous halfs of column n -> 2x ds_load_b128 ----
        v16h bf;
        const int kb = n * COLSTRIDE + kk + half * 16;
#pragma unroll
        for (int e = 0; e < 16; ++e) bf[e] = s_colT[kb + e];

        acc = __builtin_amdgcn_wmma_f32_16x16x32_f16(
            /*neg_a=*/false, a, /*neg_b=*/false, bf,
            /*c_mod=*/(short)0, acc, /*reuse_a=*/false, /*reuse_b=*/false);
    }

    // ---- Store: lane holds channel n, pixel M = r + 8*half (VGPR r of D) ----
    if (n < 3) {
#pragma unroll
        for (int r = 0; r < 8; ++r) {
            const int pix = p0 + r + 8 * half;
            float v = acc[r];
            v = fminf(fmaxf(v, 0.0f), 1.0f);
            out[((size_t)b * IMG_H * IMG_W + pix) * 3 + n] = v;
        }
    }
}

extern "C" void kernel_launch(void* const* d_in, const int* in_sizes, int n_in,
                              void* d_out, int out_size, void* d_ws, size_t ws_size,
                              hipStream_t stream) {
    const float* positions = (const float*)d_in[0];  // [B,N,2]
    const float* colors    = (const float*)d_in[1];  // [B,N,3]
    const float* sizes     = (const float*)d_in[2];  // [B,N]
    float*       out       = (float*)d_out;          // [B,H,W,3]

    dim3 grid(NBATCH * BLOCKS_PER_IMG);  // 16 * 392 = 6272 blocks
    dim3 block(256);                     // 8 waves (wave32)
    splat_wmma_kernel<<<grid, block, 0, stream>>>(positions, colors, sizes, out);
}
